// TransformerBlock_24730421691129
// MI455X (gfx1250) — matrix operand order repaired and checked by reading, not compiled, not run on hardware
//
#include <hip/hip_runtime.h>
#include <hip/hip_bf16.h>
#include <math.h>

// ---------------------------------------------------------------------------
// DiT transformer block for MI455X (gfx1250, wave32, WMMA).
// GEMMs + attention matmuls: v_wmma_f32_16x16x32_bf16 (f32 accum).
// GEMM staging: global_load_async_to_lds_b128 + double buffering (ASYNCcnt).
// ---------------------------------------------------------------------------

typedef __bf16 bf16_t;
typedef __attribute__((ext_vector_type(16))) __bf16 v16bf;
typedef __attribute__((ext_vector_type(8)))  __bf16 v8bf;
typedef __attribute__((ext_vector_type(4)))  __bf16 v4bf;
typedef __attribute__((ext_vector_type(8)))  float  v8f;

#define B_   2
#define S_   2048
#define D_   1024
#define H_   16
#define HD_  64
#define DFF_ 4096
#define EPSF 1e-6f

#define SHUF16(lo, hi) __builtin_shufflevector(lo, hi, 0,1,2,3,4,5,6,7,8,9,10,11,12,13,14,15)

static __device__ __forceinline__ v8f wmma_bf16(v16bf a, v16bf b, v8f c) {
  return __builtin_amdgcn_wmma_f32_16x16x32_bf16(false, a, false, b, (short)0, c, false, false);
}

// fast tanh via v_exp_f32 (avoids libm's branchy tanhf)
static __device__ __forceinline__ float fast_tanh(float u) {
  float e = __expf(2.f * u);
  return (e - 1.f) / (e + 1.f);
}

// async Global->LDS b128 (CDNA5 TDM-lite path, tracked by ASYNCcnt)
static __device__ __forceinline__ void async_b128(const bf16_t* g, const bf16_t* l) {
  uint32_t lo = (uint32_t)(size_t)l;       // low 32 bits of generic ptr = LDS offset
  uint64_t ga = (uint64_t)(size_t)g;
  asm volatile("global_load_async_to_lds_b128 %0, %1, off" :: "v"(lo), "v"(ga) : "memory");
}
static __device__ __forceinline__ void async_b128x2(const bf16_t* g, const bf16_t* l) {
  // INST_OFFSET applies to both LDS and global addresses -> reuse base regs
  uint32_t lo = (uint32_t)(size_t)l;
  uint64_t ga = (uint64_t)(size_t)g;
  asm volatile("global_load_async_to_lds_b128 %0, %1, off\n\t"
               "global_load_async_to_lds_b128 %0, %1, off offset:16"
               :: "v"(lo), "v"(ga) : "memory");
}
static __device__ __forceinline__ void wait_async0() {
  asm volatile("s_wait_asynccnt 0x0" ::: "memory");
}

// ---------------------------------------------------------------------------
// fp32 -> bf16 conversion (weights)
// ---------------------------------------------------------------------------
__global__ void cvt_bf16_kernel(const float* __restrict__ src, bf16_t* __restrict__ dst, int n) {
  int i = (blockIdx.x * blockDim.x + threadIdx.x) * 4;
  if (i < n) {
    float4 v = *(const float4*)(src + i);
    v4bf o; o[0] = (bf16_t)v.x; o[1] = (bf16_t)v.y; o[2] = (bf16_t)v.z; o[3] = (bf16_t)v.w;
    *(v4bf*)(dst + i) = o;
  }
}

// ---------------------------------------------------------------------------
// emb = SiLU(t) @ adaln_w^T + adaln_b      [B, 6D]  (one wave per output)
// ---------------------------------------------------------------------------
__global__ __launch_bounds__(256) void adaln_kernel(const float* __restrict__ t_,
    const float* __restrict__ w, const float* __restrict__ bias, float* __restrict__ emb) {
  int gw = blockIdx.x * 8 + (threadIdx.x >> 5);
  int lane = threadIdx.x & 31;
  int b = gw / (6 * D_);
  int j = gw % (6 * D_);
  const float* tr = t_ + (size_t)b * D_;
  const float* wr = w + (size_t)j * D_;
  float acc = 0.f;
  for (int d = lane; d < D_; d += 32) {
    float tv = tr[d];
    acc += (tv / (1.f + __expf(-tv))) * wr[d];
  }
  #pragma unroll
  for (int m = 16; m > 0; m >>= 1) acc += __shfl_xor(acc, m, 32);
  if (lane == 0) emb[(size_t)b * 6 * D_ + j] = acc + bias[j];
}

// ---------------------------------------------------------------------------
// LayerNorm + AdaLN modulate -> bf16     (one block per row)
// ---------------------------------------------------------------------------
__global__ __launch_bounds__(256) void ln_mod_kernel(const float* __restrict__ x,
    const float* __restrict__ emb, bf16_t* __restrict__ xm) {
  __shared__ float sm[16];
  int row = blockIdx.x;
  int b = row >> 11;                         // row / S_
  int t = threadIdx.x;
  const float* xr = x + (size_t)row * D_;
  float4 v = ((const float4*)xr)[t];
  float a[4] = {v.x, v.y, v.z, v.w};
  float s = a[0] + a[1] + a[2] + a[3];
  float ss = a[0]*a[0] + a[1]*a[1] + a[2]*a[2] + a[3]*a[3];
  #pragma unroll
  for (int m = 16; m > 0; m >>= 1) { s += __shfl_xor(s, m, 32); ss += __shfl_xor(ss, m, 32); }
  if ((t & 31) == 0) { sm[t >> 5] = s; sm[8 + (t >> 5)] = ss; }
  __syncthreads();
  if (t == 0) { float q = 0, r = 0; for (int i = 0; i < 8; i++) { q += sm[i]; r += sm[8+i]; } sm[0] = q; sm[8] = r; }
  __syncthreads();
  float mu = sm[0] * (1.f / D_);
  float var = sm[8] * (1.f / D_) - mu * mu;
  float rinv = rsqrtf(var + EPSF);
  const float* eb = emb + (size_t)b * 6 * D_;
  int d = t * 4;
  v4bf o;
  #pragma unroll
  for (int i = 0; i < 4; i++) {
    float ln = (a[i] - mu) * rinv;
    o[i] = (bf16_t)(ln * (1.f + eb[D_ + d + i]) + eb[d + i]);
  }
  *(v4bf*)(xm + (size_t)row * D_ + d) = o;
}

// ---------------------------------------------------------------------------
// Generic bf16 WMMA GEMM: C[M,N] = A[M,K] @ W[N,K]^T (+ fused epilogues)
// 256 thr (8 waves, 2x4), tile 64x128, K-step 32, async-LDS double buffering.
// ---------------------------------------------------------------------------
#define TM 64
#define TN 128
#define SA 40   // padded LDS row stride (bf16 units): 32 + 8
#define SB 40

enum { EPI_F32 = 0, EPI_RESID_GATE = 1, EPI_BIAS_GELU_BF16 = 2, EPI_BIAS_RESID_GATE = 3 };

__global__ __launch_bounds__(256) void gemm_bf16_kernel(
    const bf16_t* __restrict__ A, const bf16_t* __restrict__ W,
    float* __restrict__ outF, bf16_t* __restrict__ outB,
    int M, int N, int K, int mode,
    const float* __restrict__ bias, const float* __restrict__ resid,
    const float* __restrict__ gate, int gate_stride, int rows_per_b) {
  __shared__ bf16_t ldsA[2][TM * SA];
  __shared__ bf16_t ldsB[2][TN * SB];
  int m0 = blockIdx.y * TM;
  int n0 = blockIdx.x * TN;
  int tid = threadIdx.x;
  int lane = tid & 31, wave = tid >> 5;
  int wm = wave >> 2, wn = wave & 3;         // 2 x 4 wave grid
  int half = lane >> 4, r = lane & 15;

  // per-thread staging coordinates
  int idxA = tid * 8;  int am = idxA >> 5, ak = idxA & 31;   // A: 8 bf16/thread
  int idxB = tid * 16; int bn = idxB >> 5, bk = idxB & 31;   // B: 16 bf16/thread
  const bf16_t* gA = A + (size_t)(m0 + am) * K + ak;
  const bf16_t* gB = W + (size_t)(n0 + bn) * K + bk;
  bf16_t* lA0 = &ldsA[0][am * SA + ak];
  bf16_t* lA1 = &ldsA[1][am * SA + ak];
  bf16_t* lB0 = &ldsB[0][bn * SB + bk];
  bf16_t* lB1 = &ldsB[1][bn * SB + bk];

  v8f acc[2][2] = {};

  // prologue: stage first K tile into buffer 0
  async_b128(gA, lA0);
  async_b128x2(gB, lB0);

  int buf = 0;
  for (int k0 = 0; k0 < K; k0 += 32) {
    wait_async0();        // this wave's staged tile has landed in LDS
    __syncthreads();      // everyone done reading the alternate buffer + data visible
    if (k0 + 32 < K) {    // stream next tile into the other buffer
      async_b128(gA + k0 + 32, buf ? lA0 : lA1);
      async_b128x2(gB + k0 + 32, buf ? lB0 : lB1);
    }

    const bf16_t* bufA = ldsA[buf];
    const bf16_t* bufB = ldsB[buf];
    v16bf af[2], bfr[2];
    #pragma unroll
    for (int i = 0; i < 2; i++) {            // A fragment: lane-half K striping
      const bf16_t* p = &bufA[(wm * 32 + i * 16 + r) * SA + half * 8];
      af[i] = SHUF16(*(const v8bf*)p, *(const v8bf*)(p + 16));
    }
    #pragma unroll
    for (int j = 0; j < 2; j++) {            // B fragment: lane-half K striping
      const bf16_t* p = &bufB[(wn * 32 + j * 16 + r) * SB + half * 8];
      bfr[j] = SHUF16(*(const v8bf*)p, *(const v8bf*)(p + 16));
    }
    #pragma unroll
    for (int i = 0; i < 2; i++)
      #pragma unroll
      for (int j = 0; j < 2; j++)
        acc[i][j] = wmma_bf16(af[i], bfr[j], acc[i][j]);
    buf ^= 1;
  }

  // epilogue (C/D layout: VGPR e -> row e + 8*half, col = lane&15)
  #pragma unroll
  for (int i = 0; i < 2; i++)
    #pragma unroll
    for (int j = 0; j < 2; j++) {
      int rbase = m0 + wm * 32 + i * 16 + half * 8;
      int c = n0 + wn * 32 + j * 16 + r;
      #pragma unroll
      for (int e = 0; e < 8; e++) {
        int row = rbase + e;
        size_t off = (size_t)row * N + c;
        float v = acc[i][j][e];
        if (mode == EPI_F32) {
          outF[off] = v;
        } else if (mode == EPI_RESID_GATE) {
          int b = row / rows_per_b;
          outF[off] = resid[off] + gate[(size_t)b * gate_stride + c] * v;
        } else if (mode == EPI_BIAS_GELU_BF16) {
          float xv = v + bias[c];
          float g = 0.5f * xv *
                    (1.f + fast_tanh(0.7978845608028654f * (xv + 0.044715f * xv * xv * xv)));
          outB[off] = (bf16_t)g;
        } else { // EPI_BIAS_RESID_GATE
          int b = row / rows_per_b;
          float xv = v + bias[c];
          outF[off] = resid[off] + gate[(size_t)b * gate_stride + c] * xv;
        }
      }
    }
}

// ---------------------------------------------------------------------------
// RMSNorm(q/k) + RoPE + repack: qh/kh [B,H,S,64] bf16 (q pre-scaled by 1/8),
// v transposed -> vt [B,H,64,S] bf16.  One block per (b,s) row.
// ---------------------------------------------------------------------------
__global__ __launch_bounds__(256) void qk_rope_pack_kernel(
    const float* __restrict__ qf, const float* __restrict__ kf, const float* __restrict__ vf,
    const float* __restrict__ qw, const float* __restrict__ kw, const float* __restrict__ rope,
    bf16_t* __restrict__ qh, bf16_t* __restrict__ kh, bf16_t* __restrict__ vt) {
  __shared__ float sm[16];
  int row = blockIdx.x;
  int b = row >> 11, s = row & 2047;
  int t = threadIdx.x;
  float4 q4 = ((const float4*)(qf + (size_t)row * D_))[t];
  float4 k4 = ((const float4*)(kf + (size_t)row * D_))[t];
  float4 v4 = ((const float4*)(vf + (size_t)row * D_))[t];
  float qa[4] = {q4.x, q4.y, q4.z, q4.w};
  float ka[4] = {k4.x, k4.y, k4.z, k4.w};
  float va[4] = {v4.x, v4.y, v4.z, v4.w};
  float qss = qa[0]*qa[0] + qa[1]*qa[1] + qa[2]*qa[2] + qa[3]*qa[3];
  float kss = ka[0]*ka[0] + ka[1]*ka[1] + ka[2]*ka[2] + ka[3]*ka[3];
  #pragma unroll
  for (int m = 16; m > 0; m >>= 1) { qss += __shfl_xor(qss, m, 32); kss += __shfl_xor(kss, m, 32); }
  if ((t & 31) == 0) { sm[t >> 5] = qss; sm[8 + (t >> 5)] = kss; }
  __syncthreads();
  if (t == 0) { float a = 0, c = 0; for (int i = 0; i < 8; i++) { a += sm[i]; c += sm[8+i]; } sm[0] = a; sm[8] = c; }
  __syncthreads();
  float qr = rsqrtf(sm[0] * (1.f / D_) + EPSF);
  float kr = rsqrtf(sm[8] * (1.f / D_) + EPSF);

  int d = t * 4, hh = d >> 6, hd = d & 63;
  float qn[4], kn[4];
  #pragma unroll
  for (int i = 0; i < 4; i++) { qn[i] = qa[i] * qr * qw[d + i]; kn[i] = ka[i] * kr * kw[d + i]; }
  float qo[4], ko[4];
  #pragma unroll
  for (int p = 0; p < 2; p++) {
    const float* rp = rope + ((size_t)s * HD_ + hd + 2 * p) * 2;
    float c0 = rp[0], s0 = rp[1], c1 = rp[2], s1 = rp[3];
    qo[2*p]   = qn[2*p]   * c0 - qn[2*p+1] * s0;
    qo[2*p+1] = qn[2*p+1] * c1 + qn[2*p]   * s1;
    ko[2*p]   = kn[2*p]   * c0 - kn[2*p+1] * s0;
    ko[2*p+1] = kn[2*p+1] * c1 + kn[2*p]   * s1;
  }
  size_t base = (((size_t)b * H_ + hh) * S_ + s) * HD_ + hd;
  v4bf qv, kv;
  #pragma unroll
  for (int i = 0; i < 4; i++) { qv[i] = (bf16_t)(qo[i] * 0.125f); kv[i] = (bf16_t)ko[i]; }
  *(v4bf*)(qh + base) = qv;
  *(v4bf*)(kh + base) = kv;
  #pragma unroll
  for (int i = 0; i < 4; i++)
    vt[(((size_t)b * H_ + hh) * HD_ + hd + i) * S_ + s] = (bf16_t)va[i];
}

// ---------------------------------------------------------------------------
// Flash attention: one wave per 16 query rows. Key tiles of 32.
// QK^T (4 wmma) -> online softmax (shfl within lane-halves) -> P via LDS
// (A-fragment restriping) -> PV (4 wmma).
// ---------------------------------------------------------------------------
#define SP 40
__global__ __launch_bounds__(32) void attn_kernel(
    const bf16_t* __restrict__ qh, const bf16_t* __restrict__ kh,
    const bf16_t* __restrict__ vt, bf16_t* __restrict__ o) {
  __shared__ bf16_t P[16 * SP];
  int blk = blockIdx.x;
  int qt = blk & 127, bh = blk >> 7;
  int b = bh >> 4, h = bh & 15;
  int lane = threadIdx.x, half = lane >> 4, r = lane & 15;

  const bf16_t* qbase = qh + ((size_t)bh * S_ + qt * 16) * HD_;
  const bf16_t* kbase = kh + (size_t)bh * S_ * HD_;
  const bf16_t* vbase = vt + (size_t)bh * HD_ * S_;

  v16bf qfrg[2];
  #pragma unroll
  for (int c = 0; c < 2; c++) {
    const bf16_t* p = qbase + (size_t)r * HD_ + c * 32 + half * 8;
    qfrg[c] = SHUF16(*(const v8bf*)p, *(const v8bf*)(p + 16));
  }

  v8f oacc[4] = {};
  float mrun[8], lrun[8];
  #pragma unroll
  for (int i = 0; i < 8; i++) { mrun[i] = -__builtin_inff(); lrun[i] = 0.f; }

  for (int kt = 0; kt < S_; kt += 32) {
    v8f sc[2] = {};
    #pragma unroll
    for (int j = 0; j < 2; j++)
      #pragma unroll
      for (int c = 0; c < 2; c++) {
        const bf16_t* p = kbase + (size_t)(kt + j * 16 + r) * HD_ + c * 32 + half * 8;
        v16bf kf = SHUF16(*(const v8bf*)p, *(const v8bf*)(p + 16));
        sc[j] = wmma_bf16(qfrg[c], kf, sc[j]);
      }

    float mnew[8], corr[8];
    #pragma unroll
    for (int i = 0; i < 8; i++) {
      float m = fmaxf(sc[0][i], sc[1][i]);
      #pragma unroll
      for (int x = 8; x > 0; x >>= 1) m = fmaxf(m, __shfl_xor(m, x, 32));
      mnew[i] = fmaxf(mrun[i], m);
      corr[i] = __expf(mrun[i] - mnew[i]);
      mrun[i] = mnew[i];
    }
    #pragma unroll
    for (int i = 0; i < 8; i++) {
      float p0 = __expf(sc[0][i] - mnew[i]);
      float p1 = __expf(sc[1][i] - mnew[i]);
      sc[0][i] = p0; sc[1][i] = p1;
      float rs = p0 + p1;
      #pragma unroll
      for (int x = 8; x > 0; x >>= 1) rs += __shfl_xor(rs, x, 32);
      lrun[i] = lrun[i] * corr[i] + rs;
    }
    #pragma unroll
    for (int d = 0; d < 4; d++)
      #pragma unroll
      for (int i = 0; i < 8; i++) oacc[d][i] *= corr[i];

    // restripe P through LDS to A-fragment layout
    #pragma unroll
    for (int j = 0; j < 2; j++)
      #pragma unroll
      for (int i = 0; i < 8; i++)
        P[(i + 8 * half) * SP + j * 16 + r] = (bf16_t)sc[j][i];
    __syncthreads();
    const bf16_t* pp = &P[r * SP + half * 8];
    v16bf pf = SHUF16(*(const v8bf*)pp, *(const v8bf*)(pp + 16));

    #pragma unroll
    for (int d = 0; d < 4; d++) {
      const bf16_t* vp = vbase + (size_t)(d * 16 + r) * S_ + kt + half * 8;
      v16bf vfr = SHUF16(*(const v8bf*)vp, *(const v8bf*)(vp + 16));
      oacc[d] = wmma_bf16(pf, vfr, oacc[d]);
    }
    __syncthreads();
  }

  #pragma unroll
  for (int d = 0; d < 4; d++)
    #pragma unroll
    for (int i = 0; i < 8; i++) {
      int srow = qt * 16 + i + 8 * half;
      int col = h * HD_ + d * 16 + r;
      o[((size_t)b * S_ + srow) * (H_ * HD_) + col] = (bf16_t)(oacc[d][i] / lrun[i]);
    }
}

// ---------------------------------------------------------------------------
// RMSNorm + AdaLN MLP modulate -> bf16
// ---------------------------------------------------------------------------
__global__ __launch_bounds__(256) void rmsmod_kernel(const float* __restrict__ h,
    const float* __restrict__ emb, const float* __restrict__ w3, bf16_t* __restrict__ nm) {
  __shared__ float sm[8];
  int row = blockIdx.x;
  int b = row >> 11;
  int t = threadIdx.x;
  float4 v = ((const float4*)(h + (size_t)row * D_))[t];
  float a[4] = {v.x, v.y, v.z, v.w};
  float ss = a[0]*a[0] + a[1]*a[1] + a[2]*a[2] + a[3]*a[3];
  #pragma unroll
  for (int m = 16; m > 0; m >>= 1) ss += __shfl_xor(ss, m, 32);
  if ((t & 31) == 0) sm[t >> 5] = ss;
  __syncthreads();
  if (t == 0) { float q = 0; for (int i = 0; i < 8; i++) q += sm[i]; sm[0] = q; }
  __syncthreads();
  float rinv = rsqrtf(sm[0] * (1.f / D_) + EPSF);
  const float* eb = emb + (size_t)b * 6 * D_;
  int d = t * 4;
  v4bf o;
  #pragma unroll
  for (int i = 0; i < 4; i++) {
    float n3 = a[i] * rinv * w3[d + i];
    o[i] = (bf16_t)(n3 * (1.f + eb[4 * D_ + d + i]) + eb[3 * D_ + d + i]);
  }
  *(v4bf*)(nm + (size_t)row * D_ + d) = o;
}

// ---------------------------------------------------------------------------
// Host launcher
// ---------------------------------------------------------------------------
extern "C" void kernel_launch(void* const* d_in, const int* in_sizes, int n_in,
                              void* d_out, int out_size, void* d_ws, size_t ws_size,
                              hipStream_t stream) {
  const float* x        = (const float*)d_in[0];
  const float* ts       = (const float*)d_in[1];
  const float* rope     = (const float*)d_in[2];
  const float* adaln_w  = (const float*)d_in[3];
  const float* adaln_b  = (const float*)d_in[4];
  const float* wq       = (const float*)d_in[5];
  const float* wk       = (const float*)d_in[6];
  const float* wv       = (const float*)d_in[7];
  const float* wo       = (const float*)d_in[8];
  const float* q_norm_w = (const float*)d_in[9];
  const float* k_norm_w = (const float*)d_in[10];
  const float* norm3_w  = (const float*)d_in[11];
  const float* fc1_w    = (const float*)d_in[12];
  const float* fc1_bias = (const float*)d_in[13];
  const float* fc2_w    = (const float*)d_in[14];
  const float* fc2_bias = (const float*)d_in[15];
  (void)in_sizes; (void)n_in; (void)out_size; (void)ws_size;

  char* ws = (char*)d_ws;
  size_t off = 0;
  auto alloc = [&](size_t bytes) -> void* {
    void* p = ws + off;
    off += (bytes + 255) & ~(size_t)255;
    return p;
  };

  const size_t NROW = (size_t)B_ * S_;                // 4096
  bf16_t* wq_b  = (bf16_t*)alloc((size_t)D_ * D_ * 2);
  bf16_t* wk_b  = (bf16_t*)alloc((size_t)D_ * D_ * 2);
  bf16_t* wv_b  = (bf16_t*)alloc((size_t)D_ * D_ * 2);
  bf16_t* wo_b  = (bf16_t*)alloc((size_t)D_ * D_ * 2);
  bf16_t* fc1_b16 = (bf16_t*)alloc((size_t)DFF_ * D_ * 2);
  bf16_t* fc2_b16 = (bf16_t*)alloc((size_t)D_ * DFF_ * 2);
  float*  emb   = (float*)alloc((size_t)B_ * 6 * D_ * 4);
  bf16_t* xm    = (bf16_t*)alloc(NROW * D_ * 2);      // later reused as o_bf16
  float*  q_f   = (float*)alloc(NROW * D_ * 4);       // later reused as h
  float*  k_f   = (float*)alloc(NROW * D_ * 4);       // k_f..v_f reused as ff1 (32MB)
  float*  v_f   = (float*)alloc(NROW * D_ * 4);
  bf16_t* qh    = (bf16_t*)alloc(NROW * D_ * 2);      // later reused as nm
  bf16_t* kh    = (bf16_t*)alloc(NROW * D_ * 2);
  bf16_t* vt    = (bf16_t*)alloc(NROW * D_ * 2);
  bf16_t* o_bf  = xm;
  float*  h_f   = q_f;
  bf16_t* ff1   = (bf16_t*)k_f;
  bf16_t* nm    = qh;

  // 1) weights -> bf16
  auto cvt = [&](const float* s, bf16_t* dst, int n) {
    cvt_bf16_kernel<<<(n / 4 + 255) / 256, 256, 0, stream>>>(s, dst, n);
  };
  cvt(wq, wq_b, D_ * D_);
  cvt(wk, wk_b, D_ * D_);
  cvt(wv, wv_b, D_ * D_);
  cvt(wo, wo_b, D_ * D_);
  cvt(fc1_w, fc1_b16, DFF_ * D_);
  cvt(fc2_w, fc2_b16, D_ * DFF_);

  // 2) AdaLN embedding
  adaln_kernel<<<(B_ * 6 * D_) / 8, 256, 0, stream>>>(ts, adaln_w, adaln_b, emb);

  // 3) LayerNorm + modulate -> xm (bf16)
  ln_mod_kernel<<<NROW, 256, 0, stream>>>(x, emb, xm);

  // 4) QKV projections (WMMA)
  dim3 gD(D_ / TN, NROW / TM);
  gemm_bf16_kernel<<<gD, 256, 0, stream>>>(xm, wq_b, q_f, nullptr, NROW, D_, D_,
                                           EPI_F32, nullptr, nullptr, nullptr, 0, S_);
  gemm_bf16_kernel<<<gD, 256, 0, stream>>>(xm, wk_b, k_f, nullptr, NROW, D_, D_,
                                           EPI_F32, nullptr, nullptr, nullptr, 0, S_);
  gemm_bf16_kernel<<<gD, 256, 0, stream>>>(xm, wv_b, v_f, nullptr, NROW, D_, D_,
                                           EPI_F32, nullptr, nullptr, nullptr, 0, S_);

  // 5) RMSNorm + RoPE + repack (q pre-scaled by HD^-0.5)
  qk_rope_pack_kernel<<<NROW, 256, 0, stream>>>(q_f, k_f, v_f, q_norm_w, k_norm_w,
                                                rope, qh, kh, vt);

  // 6) Flash attention (WMMA) -> o_bf (reuses xm)
  attn_kernel<<<B_ * H_ * (S_ / 16), 32, 0, stream>>>(qh, kh, vt, o_bf);

  // 7) O projection + gated residual: h = x + g_msa * (o @ wo^T)
  gemm_bf16_kernel<<<gD, 256, 0, stream>>>(o_bf, wo_b, h_f, nullptr, NROW, D_, D_,
                                           EPI_RESID_GATE, nullptr, x, emb + 2 * D_,
                                           6 * D_, S_);

  // 8) RMSNorm + MLP modulate -> nm (bf16, reuses qh)
  rmsmod_kernel<<<NROW, 256, 0, stream>>>(h_f, emb, norm3_w, nm);

  // 9) FC1 + bias + GELU -> ff1 (bf16, reuses k_f/v_f)
  dim3 gF1(DFF_ / TN, NROW / TM);
  gemm_bf16_kernel<<<gF1, 256, 0, stream>>>(nm, fc1_b16, nullptr, ff1, NROW, DFF_, D_,
                                            EPI_BIAS_GELU_BF16, fc1_bias, nullptr,
                                            nullptr, 0, S_);

  // 10) FC2 + bias + gated residual -> out = h + g_mlp * ff
  gemm_bf16_kernel<<<gD, 256, 0, stream>>>(ff1, fc2_b16, (float*)d_out, nullptr, NROW,
                                           D_, DFF_, EPI_BIAS_RESID_GATE, fc2_bias,
                                           h_f, emb + 5 * D_, 6 * D_, S_);
}
